// EdgeAwareBiLSTM_35862976921870
// MI455X (gfx1250) — matrix operand order repaired and checked by reading, not compiled, not run on hardware
//
#include <hip/hip_runtime.h>
#include <hip/hip_bf16.h>
#include <stdint.h>

#define B_ 32
#define C_ 64
#define L_ 4096
#define H_ 32
#define G4 128   // 4*H

typedef __attribute__((ext_vector_type(16))) __bf16 v16bf;
typedef __attribute__((ext_vector_type(8)))  float  v8f;
typedef __attribute__((ext_vector_type(4)))  unsigned int v4u;

union V16U { v16bf bf; v4u q[2]; };

// ---- WMMA (bf16 16x16x32, wave32) lane-layout helpers (ISA 7.12.2) ----
// A 16x32:  M = lane&15 ; K(e) = e + (e&8) + ((lane&16)>>1)
__device__ __forceinline__ int A_K(int e, int lane) { return e + (e & 8) + ((lane & 16) >> 1); }
// B 32x16:  N = lane&15 ; K(e) = e + (e&8) + ((lane&16)>>1)
__device__ __forceinline__ int B_K(int e, int lane) { return e + (e & 8) + ((lane & 16) >> 1); }
// C/D 16x16 f32: N = lane&15 ; M(r) = r + ((lane&16)>>1)
__device__ __forceinline__ int D_M(int r, int lane) { return r + ((lane & 16) >> 1); }

__device__ __forceinline__ float sigmoidf_(float x) { return 1.f / (1.f + __expf(-x)); }

// Load a pre-swizzled B tile: 32 lanes x 16 bf16, lane-contiguous (two b128 loads).
__device__ __forceinline__ v16bf load_sw_tile(const __bf16* tile, int lane) {
  V16U u;
  const v4u* p = (const v4u*)(tile + lane * 16);
  u.q[0] = p[0];
  u.q[1] = p[1];
  return u.bf;
}

// Load an A tile from a row-major bf16 [rows][32] array (global or LDS):
// lane reads row (row0 + lane&15), 2 x 16B slices selected by lane half.
__device__ __forceinline__ v16bf load_a_tile(const __bf16* rows32, int row0, int lane) {
  const __bf16* hp = rows32 + (size_t)(row0 + (lane & 15)) * 32;
  const int hb = (lane & 16) >> 1;   // +8 bf16 for high half-lanes
  V16U u;
  u.q[0] = *(const v4u*)(hp + hb);        // K = 8*hi .. 8*hi+7
  u.q[1] = *(const v4u*)(hp + 16 + hb);   // K = 16+8*hi .. 16+8*hi+7
  return u.bf;
}

// Async global -> LDS copy, 16B per thread (ASYNCcnt-tracked, ISA 10.7/15.18).
__device__ __forceinline__ void async_copy_b128(unsigned lds_off, const float* src) {
  asm volatile("global_load_async_to_lds_b128 %0, %1, off"
               :: "v"(lds_off), "v"(src) : "memory");
}
__device__ __forceinline__ void wait_async0() {
  asm volatile("s_wait_asynccnt 0x0" ::: "memory");
}

// ---------------------------------------------------------------------------
// K0: fold projection into LSTM input weights (W = w_proj @ wih_d), and write
//     ALL B-operand weights pre-swizzled into WMMA lane order (lane-contiguous
//     16 bf16 per lane per tile): Wsw[2][2kt][8nt], whhsw[2][8nt], w1sw[2kt][2nt].
// ---------------------------------------------------------------------------
__global__ void k_prep(const float* __restrict__ w_proj, const float* __restrict__ b_proj,
                       const float* __restrict__ wih_f, const float* __restrict__ whh_f,
                       const float* __restrict__ b_f,
                       const float* __restrict__ wih_b, const float* __restrict__ whh_b,
                       const float* __restrict__ b_b,
                       const float* __restrict__ w1,
                       __bf16* __restrict__ Wsw, __bf16* __restrict__ whhsw,
                       __bf16* __restrict__ w1sw, float* __restrict__ biasf)
{
  const int t = threadIdx.x;
  // folded input weights, swizzled: idx = (((d*2+kt)*8+nt)*32+lane)*16+e
  for (int idx = t; idx < 2 * 2 * 8 * 32 * 16; idx += blockDim.x) {
    int e = idx & 15, lane = (idx >> 4) & 31, nt = (idx >> 9) & 7;
    int kt = (idx >> 12) & 1, d = (idx >> 13) & 1;
    int k = kt * 32 + B_K(e, lane);            // input channel 0..63
    int n = nt * 16 + (lane & 15);             // gate column 0..127
    const float* wih = d ? wih_b : wih_f;
    float s = 0.f;
    for (int h = 0; h < H_; ++h) s += w_proj[k * H_ + h] * wih[h * G4 + n];
    Wsw[idx] = (__bf16)s;
  }
  // folded bias
  for (int idx = t; idx < 2 * G4; idx += blockDim.x) {
    int d = idx / G4, g = idx % G4;
    const float* wih = d ? wih_b : wih_f;
    const float* bb  = d ? b_b : b_f;
    float s = bb[g];
    for (int h = 0; h < H_; ++h) s += b_proj[h] * wih[h * G4 + g];
    biasf[idx] = s;
  }
  // recurrent weights, swizzled: idx = ((d*8+nt)*32+lane)*16+e
  for (int idx = t; idx < 2 * 8 * 32 * 16; idx += blockDim.x) {
    int e = idx & 15, lane = (idx >> 4) & 31, nt = (idx >> 9) & 7, d = (idx >> 12) & 1;
    int k = B_K(e, lane);                      // hidden 0..31
    int n = nt * 16 + (lane & 15);
    const float* whh = d ? whh_b : whh_f;
    whhsw[idx] = (__bf16)whh[k * G4 + n];
  }
  // classifier layer-1 weights, swizzled: idx = ((kt*2+nt)*32+lane)*16+e
  for (int idx = t; idx < 2 * 2 * 32 * 16; idx += blockDim.x) {
    int e = idx & 15, lane = (idx >> 4) & 31, nt = (idx >> 9) & 1, kt = (idx >> 10) & 1;
    int k = kt * 32 + B_K(e, lane);            // 0..63
    int n = nt * 16 + (lane & 15);             // 0..31
    w1sw[idx] = (__bf16)w1[k * H_ + n];
  }
}

// ---------------------------------------------------------------------------
// K1: all-timestep input gates Gin[d][l][b][0:128] = x[b,l,:] @ W[d] + bias[d]
//     GEMM [B*L, 64] x [64, 128] via WMMA bf16. 8 waves/block, 16 rows/wave.
// ---------------------------------------------------------------------------
__global__ void __launch_bounds__(256)
k_gates(const float* __restrict__ features,
        const __bf16* __restrict__ Wsw,
        const float* __restrict__ biasf,
        float* __restrict__ Gin)
{
  const int lane = threadIdx.x & 31;
  const int wv   = threadIdx.x >> 5;                 // 8 waves
  const int bid  = blockIdx.x;
  const int d    = bid / (B_ * (L_ / 128));
  const int rem  = bid % (B_ * (L_ / 128));
  const int b    = rem / (L_ / 128);
  const int lc   = rem % (L_ / 128);
  const int l0   = lc * 128 + wv * 16;

  const int m = lane & 15;
  v16bf a0, a1;                                      // K = channel 0..31 / 32..63
#pragma unroll
  for (int e = 0; e < 16; ++e) {
    int k = A_K(e, lane);
    a0[e] = (__bf16)features[((size_t)b * C_ + k)      * L_ + (l0 + m)];
    a1[e] = (__bf16)features[((size_t)b * C_ + k + 32) * L_ + (l0 + m)];
  }
  const __bf16* W = Wsw + (size_t)d * 2 * 8 * 512;   // [kt][nt][lane][16]
  const int n16 = lane & 15;
#pragma unroll
  for (int nt = 0; nt < 8; ++nt) {
    const int gcol = nt * 16 + n16;
    v16bf bt0 = load_sw_tile(W + (0 * 8 + nt) * 512, lane);
    v16bf bt1 = load_sw_tile(W + (1 * 8 + nt) * 512, lane);
    const float bv = biasf[d * G4 + gcol];
    v8f acc = {bv, bv, bv, bv, bv, bv, bv, bv};
    acc = __builtin_amdgcn_wmma_f32_16x16x32_bf16(false, a0, false, bt0, (short)0, acc, false, false);
    acc = __builtin_amdgcn_wmma_f32_16x16x32_bf16(false, a1, false, bt1, (short)0, acc, false, false);
#pragma unroll
    for (int r = 0; r < 8; ++r) {
      int l = l0 + D_M(r, lane);
      Gin[(((size_t)d * L_ + l) * B_ + b) * G4 + gcol] = acc[r];
    }
  }
}

// ---------------------------------------------------------------------------
// K2: sequential BiLSTM scan. 2 blocks (fwd/bwd) x 512 threads (16 waves).
//     Per step: one v_wmma per wave (g = Gin + h @ whh, K=32 in one op).
//     whh B-tiles pinned in registers for all 4096 steps; h in 2 KB LDS;
//     Gin double-buffered in LDS via global_load_async_to_lds_b128
//     (ASYNCcnt), so 8 KB (16 B/thread) of step t+1's gates streams in under step t's compute.
// ---------------------------------------------------------------------------
__global__ void __launch_bounds__(512, 1)
k_scan(const float* __restrict__ Gin,
       const __bf16* __restrict__ whhsw,
       __bf16* __restrict__ hbf,
       float* __restrict__ sum_h)
{
  const int d    = blockIdx.x;
  const int tid  = threadIdx.x;
  const int lane = tid & 31;
  const int wv   = tid >> 5;      // 16 waves
  const int mh   = wv >> 3;       // batch-half tile
  const int nt   = wv & 7;        // gate-column tile

  __shared__ __align__(16) __bf16 h_buf[32][32];   // [batch][hid]    2 KB
  __shared__ float  g_buf[32][128];                // post-WMMA gates 16 KB
  __shared__ float  g_in[2][32][128];              // async dbl-buf   32 KB

  const int gcol = nt * 16 + (lane & 15);
  const v16bf bw = load_sw_tile(whhsw + ((size_t)d * 8 + nt) * 512, lane);  // loop-invariant

  for (int i = tid; i < 32 * 32; i += 512) ((__bf16*)h_buf)[i] = (__bf16)0.f;

  const int b0 = tid >> 5;   // this thread owns cells (b0, j0) and (b0+16, j0)
  const int j0 = tid & 31;
  float c0 = 0.f, c1 = 0.f, s0 = 0.f, s1 = 0.f;
  const int mrow = mh * 16;

  const unsigned lds_g0 = (unsigned)(uintptr_t)&g_in[0][0][0] + tid * 32u;
  const unsigned lds_g1 = (unsigned)(uintptr_t)&g_in[1][0][0] + tid * 32u;

  // prologue: pull step 0's gate block into buffer 0
  {
    const int lf = d ? (L_ - 1) : 0;
    async_copy_b128(lds_g0, Gin + ((size_t)d * L_ + lf) * (B_ * G4) + tid * 8);
    wait_async0();
  }
  __syncthreads();

  for (int t = 0; t < L_; ++t) {
    const int l = d ? (L_ - 1 - t) : t;
    const int buf = t & 1;
    if (t + 1 < L_) {                                // stream step t+1 into the other buffer
      const int ln = d ? (l - 1) : (l + 1);
      async_copy_b128(buf ? lds_g0 : lds_g1,
                      Gin + ((size_t)d * L_ + ln) * (B_ * G4) + tid * 8);
    }
    v8f acc;                                         // C = precomputed input gates (LDS)
#pragma unroll
    for (int r = 0; r < 8; ++r)
      acc[r] = g_in[buf][mrow + D_M(r, lane)][gcol];
    const v16bf a = load_a_tile(&h_buf[0][0], mrow, lane);   // A = h (two b128 LDS loads)
    acc = __builtin_amdgcn_wmma_f32_16x16x32_bf16(false, a, false, bw, (short)0, acc, false, false);
#pragma unroll
    for (int r = 0; r < 8; ++r)
      g_buf[mrow + D_M(r, lane)][gcol] = acc[r];
    __syncthreads();

    {
      float gi = g_buf[b0][j0], gf = g_buf[b0][32 + j0];
      float gg = g_buf[b0][64 + j0], go = g_buf[b0][96 + j0];
      c0 = sigmoidf_(gf) * c0 + sigmoidf_(gi) * tanhf(gg);
      float h = sigmoidf_(go) * tanhf(c0);
      h_buf[b0][j0] = (__bf16)h;
      hbf[(((size_t)d * L_ + l) * B_ + b0) * H_ + j0] = (__bf16)h;
      s0 += h;
    }
    {
      const int bh = b0 + 16;
      float gi = g_buf[bh][j0], gf = g_buf[bh][32 + j0];
      float gg = g_buf[bh][64 + j0], go = g_buf[bh][96 + j0];
      c1 = sigmoidf_(gf) * c1 + sigmoidf_(gi) * tanhf(gg);
      float h = sigmoidf_(go) * tanhf(c1);
      h_buf[bh][j0] = (__bf16)h;
      hbf[(((size_t)d * L_ + l) * B_ + bh) * H_ + j0] = (__bf16)h;
      s1 += h;
    }
    wait_async0();        // own 16B slice of step t+1 is in LDS
    __syncthreads();      // everyone's slice visible before next iteration reads
  }
  sum_h[((size_t)d * 32 + b0)      * 32 + j0] = s0;
  sum_h[((size_t)d * 32 + b0 + 16) * 32 + j0] = s1;
}

// ---------------------------------------------------------------------------
// K3: boundary classifier. Layer1 [B*L,64]x[64,32] via WMMA -> ReLU -> LDS,
//     then per-row dot with w2 -> sigmoid -> end boost -> clamp -> probs.
// ---------------------------------------------------------------------------
__global__ void __launch_bounds__(256)
k_cls(const __bf16* __restrict__ hbf,
      const __bf16* __restrict__ w1sw,
      const float* __restrict__ b1,
      const float* __restrict__ w2,
      const float* __restrict__ b2,
      float* __restrict__ out_probs)
{
  const int lane = threadIdx.x & 31;
  const int wv   = threadIdx.x >> 5;     // 8 waves
  const int b    = blockIdx.x / (L_ / 128);
  const int lc   = blockIdx.x % (L_ / 128);
  const int l0b  = lc * 128;
  const int l0   = l0b + wv * 16;

  __shared__ float hcls[128][H_];        // 16 KB

  // A tiles: rows = 16 consecutive l (for this b); K = fwd h then bwd h.
  // hbf rows for fixed b stride by B_*H_ bf16; build per-wave row pointers.
  const int m = lane & 15;
  const __bf16* rf = hbf + (((size_t)0 * L_ + l0 + m) * B_ + b) * H_;
  const __bf16* rb = hbf + (((size_t)1 * L_ + l0 + m) * B_ + b) * H_;
  const int hb = (lane & 16) >> 1;
  V16U ua0, ua1;
  ua0.q[0] = *(const v4u*)(rf + hb);      ua0.q[1] = *(const v4u*)(rf + 16 + hb);
  ua1.q[0] = *(const v4u*)(rb + hb);      ua1.q[1] = *(const v4u*)(rb + 16 + hb);

#pragma unroll
  for (int nt = 0; nt < 2; ++nt) {
    const int n = nt * 16 + (lane & 15);
    v16bf bt0 = load_sw_tile(w1sw + (0 * 2 + nt) * 512, lane);
    v16bf bt1 = load_sw_tile(w1sw + (1 * 2 + nt) * 512, lane);
    const float bv = b1[n];
    v8f acc = {bv, bv, bv, bv, bv, bv, bv, bv};
    acc = __builtin_amdgcn_wmma_f32_16x16x32_bf16(false, ua0.bf, false, bt0, (short)0, acc, false, false);
    acc = __builtin_amdgcn_wmma_f32_16x16x32_bf16(false, ua1.bf, false, bt1, (short)0, acc, false, false);
#pragma unroll
    for (int r = 0; r < 8; ++r)
      hcls[wv * 16 + D_M(r, lane)][n] = fmaxf(acc[r], 0.f);
  }
  __syncthreads();

  const int t = threadIdx.x;
  if (t < 128) {
    float s = b2[0];
#pragma unroll
    for (int j = 0; j < H_; ++j) s += hcls[t][j] * w2[j];
    float p = sigmoidf_(s);
    const int l = l0b + t;
    if (l == 0 || l == L_ - 1) p *= 1.2f;
    p = fminf(fmaxf(p, 0.f), 1.f);
    out_probs[(size_t)b * L_ + l] = p;
  }
}

// ---------------------------------------------------------------------------
// K4: boundary adjustment from clamped probs + real-classifier head.
// ---------------------------------------------------------------------------
__global__ void k_fin(float* __restrict__ out,          // [probs | adj | real]
                      const float* __restrict__ sum_h,
                      const float* __restrict__ wr,
                      const float* __restrict__ br)
{
  const int gid = blockIdx.x * blockDim.x + threadIdx.x;
  if (gid < B_ * L_) {
    const int b = gid / L_, i = gid % L_;
    const float* p = out + (size_t)b * L_;
    float adj = 0.f;
    if (i >= 1 && i <= L_ - 2) {
      float lg = p[i] - p[i - 1];
      float rg = p[i] - p[i + 1];
      if (lg < 0.f && fabsf(lg) > fabsf(rg)) adj = -1.f;
      else if (rg < 0.f && fabsf(rg) > fabsf(lg)) adj = 1.f;
    }
    out[(size_t)B_ * L_ + gid] = adj;
  }
  if (gid < B_) {
    float s = br[0];
#pragma unroll
    for (int j = 0; j < 2 * H_; ++j) {
      const int d = j >> 5;
      float mean = sum_h[((size_t)d * 32 + gid) * 32 + (j & 31)] * (1.f / L_);
      s += mean * wr[j];
    }
    out[(size_t)2 * B_ * L_ + gid] = sigmoidf_(s);
  }
}

// ---------------------------------------------------------------------------
extern "C" void kernel_launch(void* const* d_in, const int* in_sizes, int n_in,
                              void* d_out, int out_size, void* d_ws, size_t ws_size,
                              hipStream_t stream)
{
  (void)in_sizes; (void)n_in; (void)out_size; (void)ws_size;
  const float* features = (const float*)d_in[0];
  const float* w_proj   = (const float*)d_in[1];
  const float* b_proj   = (const float*)d_in[2];
  const float* wih_f    = (const float*)d_in[3];
  const float* whh_f    = (const float*)d_in[4];
  const float* b_f      = (const float*)d_in[5];
  const float* wih_b    = (const float*)d_in[6];
  const float* whh_b    = (const float*)d_in[7];
  const float* b_b      = (const float*)d_in[8];
  const float* w1       = (const float*)d_in[9];
  const float* b1       = (const float*)d_in[10];
  const float* w2       = (const float*)d_in[11];
  const float* b2       = (const float*)d_in[12];
  const float* wr       = (const float*)d_in[13];
  const float* br       = (const float*)d_in[14];
  float* out = (float*)d_out;

  char* ws = (char*)d_ws;                            // workspace layout (bytes)
  __bf16* Wsw   = (__bf16*)(ws + 0);                 //   32768  folded W, swizzled [2][2][8][32][16]
  __bf16* whhsw = (__bf16*)(ws + 32768);             //   16384  whh, swizzled [2][8][32][16]
  __bf16* w1sw  = (__bf16*)(ws + 49152);             //    4096  w1, swizzled [2][2][32][16]
  float*  biasf = (float*) (ws + 53248);             //    1024  folded bias [2][128]
  float*  sum_h = (float*) (ws + 54272);             //    8192  temporal sums [2][32][32]
  __bf16* hbf   = (__bf16*)(ws + 62464);             // 16.8 MB  hidden states bf16 [2][L][B][32]
  float*  Gin   = (float*) (ws + 16839680);          //  134 MB  input gates f32 [2][L][B][128]

  k_prep <<<1, 256, 0, stream>>>(w_proj, b_proj, wih_f, whh_f, b_f,
                                 wih_b, whh_b, b_b, w1, Wsw, whhsw, w1sw, biasf);
  k_gates<<<2 * B_ * (L_ / 128), 256, 0, stream>>>(features, Wsw, biasf, Gin);
  k_scan <<<2, 512, 0, stream>>>(Gin, whhsw, hbf, sum_h);
  k_cls  <<<B_ * (L_ / 128), 256, 0, stream>>>(hbf, w1sw, b1, w2, b2, out);
  k_fin  <<<(B_ * L_ + 255) / 256, 256, 0, stream>>>(out, sum_h, wr, br);
}
